// QDGAT_33775622815882
// MI455X (gfx1250) — compile-verified, matrix operand order checked by reading, not run on hardware
//
#include <hip/hip_runtime.h>
#include <hip/hip_bf16.h>
#include <stdint.h>

// ---------------------------------------------------------------------------
// QDGAT forward for MI455X (gfx1250, wave32, WMMA).
// Big GEMMs: bf16 operands via v_wmma_f32_16x16x32_bf16 (f32 accumulate),
// with CDNA5 GLOBAL_LOAD_ASYNC_TO_LDS_B128 double-buffered staging.
// Softmax / small linears stay f32.
// ---------------------------------------------------------------------------

typedef __attribute__((ext_vector_type(16))) __bf16 v16bf;
typedef __attribute__((ext_vector_type(8)))  __bf16 v8bf;
typedef __attribute__((ext_vector_type(8)))  float  v8f;

__device__ __forceinline__ __bf16 f2bf(float f) {
  union { float f; uint32_t u; } x; x.f = f;
  uint32_t r = x.u + 0x7FFFu + ((x.u >> 16) & 1u);   // round-to-nearest-even
  union { uint16_t u; __bf16 b; } y; y.u = (uint16_t)(r >> 16);
  return y.b;
}

// CDNA5 async copy: 16B per lane, global -> LDS, tracked by ASYNCcnt.
__device__ __forceinline__ void async_ld16(uint32_t ldsOff, const __bf16* gp) {
  asm volatile("global_load_async_to_lds_b128 %0, %1, off"
               :: "v"(ldsOff), "v"((uint64_t)(uintptr_t)gp)
               : "memory");
}
__device__ __forceinline__ void wait_async_le4() {
  asm volatile("s_wait_asynccnt 4" ::: "memory");
}
__device__ __forceinline__ void wait_async_0() {
  asm volatile("s_wait_asynccnt 0" ::: "memory");
}

// ---------------------------------------------------------------------------
// Generic NT GEMM:  C[m,n] = (sum_k A[m,k] * W[n,k] + bias[n]) * scale[batch,n]
// A: bf16 [M,K] row-major (lda), W: bf16 [N,K] row-major (ldb, + batch stride)
// Block tile 128x128, 8 waves, wave tile 64x32, K-step 32, async double buffer.
// ---------------------------------------------------------------------------
struct GemmArgs {
  const __bf16* A;  int lda;
  const __bf16* B;  int ldb;  long long strideB;   // per-batch (1024 rows of A)
  const float*  bias;
  const float*  scale; int scaleStride;            // scale[batch*scaleStride+n]
  float*  outF; int ldcF;                          // optional f32 out
  __bf16* outB; int ldcB;                          // optional bf16 out
  int K;
};

#define LDS_STRIDE 40   // 32 k-elements + 8 pad (16B-aligned rows, bank spread)

__global__ __launch_bounds__(256) void k_gemm_bf16(GemmArgs g) {
  __shared__ __bf16 lA[2][128 * LDS_STRIDE];
  __shared__ __bf16 lB[2][128 * LDS_STRIDE];

  const int tid  = threadIdx.x;
  const int lane = tid & 31;
  const int w    = tid >> 5;
  const int wm   = w >> 2;        // 0..1  (64-row wave tile)
  const int wn   = w & 3;         // 0..3  (32-col wave tile)
  const int n0   = blockIdx.x * 128;
  const int m0   = blockIdx.y * 128;
  const int batch = m0 >> 10;     // 1024 rows per batch

  const __bf16* Bbase = g.B + (long long)batch * g.strideB;

  // Loader: thread t<128 stages A row m0+t; t>=128 stages W row n0+(t-128).
  const __bf16* grow;
  uint32_t lrow0, lrow1;          // LDS byte offsets of this thread's row
  if (tid < 128) {
    grow  = g.A + (long long)(m0 + tid) * g.lda;
    lrow0 = (uint32_t)(uintptr_t)&lA[0][tid * LDS_STRIDE];
    lrow1 = (uint32_t)(uintptr_t)&lA[1][tid * LDS_STRIDE];
  } else {
    const int r = tid - 128;
    grow  = Bbase + (long long)(n0 + r) * g.ldb;
    lrow0 = (uint32_t)(uintptr_t)&lB[0][r * LDS_STRIDE];
    lrow1 = (uint32_t)(uintptr_t)&lB[1][r * LDS_STRIDE];
  }

  const int KT = g.K >> 5;

  // Issue one 64B row slice (4 x async b128) for tile kt into buffer kt&1.
  auto issue = [&](int kt) {
    const __bf16* src = grow + kt * 32;
    const uint32_t dst = (kt & 1) ? lrow1 : lrow0;
    async_ld16(dst,      src);
    async_ld16(dst + 16, src + 8);
    async_ld16(dst + 32, src + 16);
    async_ld16(dst + 48, src + 24);
  };

  issue(0);
  if (KT > 1) issue(1);

  v8f acc[4][2];
#pragma unroll
  for (int a = 0; a < 4; ++a)
#pragma unroll
    for (int b = 0; b < 2; ++b)
#pragma unroll
      for (int i = 0; i < 8; ++i) acc[a][b][i] = 0.0f;

  const int row = lane & 15;
  const int ko  = (lane >> 4) << 3;   // 0 or 8 : bf16 WMMA per-lane K split

  for (int kt = 0; kt < KT; ++kt) {
    // Ensure tile kt has landed: async loads complete in order, and at most
    // the 4 ops for tile kt+1 may still be outstanding.
    if (kt + 1 < KT) wait_async_le4(); else wait_async_0();
    __syncthreads();   // every wave's slice of tile kt is now in LDS

    const __bf16* bufA = lA[kt & 1];
    const __bf16* bufB = lB[kt & 1];

    v16bf af[4];
#pragma unroll
    for (int fm = 0; fm < 4; ++fm) {
      const __bf16* p = bufA + (wm * 64 + fm * 16 + row) * LDS_STRIDE + ko;
      union { v16bf v; struct { v8bf lo, hi; } s; } u;
      u.s.lo = *(const v8bf*)p;
      u.s.hi = *(const v8bf*)(p + 16);
      af[fm] = u.v;
    }
#pragma unroll
    for (int fn = 0; fn < 2; ++fn) {
      const __bf16* p = bufB + (wn * 32 + fn * 16 + row) * LDS_STRIDE + ko;
      union { v16bf v; struct { v8bf lo, hi; } s; } u;
      u.s.lo = *(const v8bf*)p;
      u.s.hi = *(const v8bf*)(p + 16);
      v16bf bfr = u.v;
#pragma unroll
      for (int fm = 0; fm < 4; ++fm)
        acc[fm][fn] = __builtin_amdgcn_wmma_f32_16x16x32_bf16(
            false, af[fm], false, bfr, (short)0, acc[fm][fn], false, false);
    }

    __syncthreads();                 // all waves done reading buffer kt&1
    if (kt + 2 < KT) issue(kt + 2);  // refill it for tile kt+2
  }

  // Epilogue. C layout: lanes 0-15 -> M=vgpr, N=lane; lanes 16-31 -> M=8+vgpr.
  const int mrel = (lane >> 4) << 3;
  const int ncol = lane & 15;
#pragma unroll
  for (int fm = 0; fm < 4; ++fm)
#pragma unroll
    for (int fn = 0; fn < 2; ++fn) {
      const int mg = m0 + wm * 64 + fm * 16 + mrel;
      const int ng = n0 + wn * 32 + fn * 16 + ncol;
      const float bv = g.bias  ? g.bias[ng] : 0.0f;
      const float sv = g.scale ? g.scale[batch * g.scaleStride + ng] : 1.0f;
      v8f c = acc[fm][fn];
#pragma unroll
      for (int i = 0; i < 8; ++i) {
        const float val = (c[i] + bv) * sv;
        if (g.outF) g.outF[(long long)(mg + i) * g.ldcF + ng] = val;
        if (g.outB) g.outB[(long long)(mg + i) * g.ldcB + ng] = f2bf(val);
      }
    }
}

// ---------------------------------------------------------------------------
// Small helper kernels
// ---------------------------------------------------------------------------

__global__ void k_cvt_bf(const float* __restrict__ src, __bf16* __restrict__ dst, long long n) {
  for (long long i = blockIdx.x * (long long)blockDim.x + threadIdx.x; i < n;
       i += (long long)gridDim.x * blockDim.x)
    dst[i] = f2bf(src[i]);
}

// x_loc bf16 -> xjoint[:,0:1024] (ld 3072) and xfin[:,0:1024] (ld 2048)
__global__ void k_init_xloc(const float* __restrict__ ents, __bf16* xjoint, __bf16* xfin) {
  for (long long i = blockIdx.x * (long long)blockDim.x + threadIdx.x; i < 4096LL * 1024;
       i += (long long)gridDim.x * blockDim.x) {
    const int m = (int)(i >> 10), d = (int)(i & 1023);
    const __bf16 v = f2bf(ents[i]);
    xjoint[(long long)m * 3072 + d] = v;
    xfin  [(long long)m * 2048 + d] = v;
  }
}

__global__ void k_init_xctx(const float* __restrict__ init_mem, float* __restrict__ xctx) {
  for (long long i = blockIdx.x * (long long)blockDim.x + threadIdx.x; i < 4096LL * 1024;
       i += (long long)gridDim.x * blockDim.x)
    xctx[i] = init_mem[i & 1023];
}

// xc = x_ctx * mask, stored bf16 into xjoint[:,1024:2048] and xcm[:,0:1024]
__global__ void k_prep_xc(const float* __restrict__ xctx, const float* __restrict__ mask,
                          __bf16* xjoint, __bf16* xcm) {
  for (long long i = blockIdx.x * (long long)blockDim.x + threadIdx.x; i < 4096LL * 1024;
       i += (long long)gridDim.x * blockDim.x) {
    const int m = (int)(i >> 10), d = (int)(i & 1023);
    const __bf16 v = f2bf(xctx[i] * mask[m]);
    xjoint[(long long)m * 3072 + 1024 + d] = v;
    xcm   [(long long)m * 2048 + d]        = v;
  }
}

// xjoint[:,2048:3072] = bf16(pl * pc)
__global__ void k_pack_joint(const float* __restrict__ pl, const float* __restrict__ pc,
                             __bf16* xjoint) {
  for (long long i = blockIdx.x * (long long)blockDim.x + threadIdx.x; i < 4096LL * 1024;
       i += (long long)gridDim.x * blockDim.x) {
    const int m = (int)(i >> 10), d = (int)(i & 1023);
    xjoint[(long long)m * 3072 + 2048 + d] = f2bf(pl[i] * pc[i]);
  }
}

// xfin[:,1024:2048] = bf16(x_ctx)   (unmasked, per reference final concat)
__global__ void k_fin_xctx(const float* __restrict__ xctx, __bf16* xfin) {
  for (long long i = blockIdx.x * (long long)blockDim.x + threadIdx.x; i < 4096LL * 1024;
       i += (long long)gridDim.x * blockDim.x) {
    const int m = (int)(i >> 10), d = (int)(i & 1023);
    xfin[(long long)m * 2048 + 1024 + d] = f2bf(xctx[i]);
  }
}

// svec [4,3072] = [1 | pk_t | pv_t]
__global__ void k_svec(const float* __restrict__ pk, const float* __restrict__ pv,
                       float* __restrict__ svec) {
  const int i = blockIdx.x * blockDim.x + threadIdx.x;
  if (i >= 4 * 3072) return;
  const int b = i / 3072, c = i % 3072;
  float v;
  if (c < 1024)      v = 1.0f;
  else if (c < 2048) v = pk[b * 1024 + (c - 1024)];
  else               v = pv[b * 1024 + (c - 2048)];
  svec[i] = v;
}

// Tiny f32 linear: out[r,n] = act( dot(X[r,:], W[n,:]) + b[n] )   (R*N threads)
__global__ void k_small_linear(const float* __restrict__ X, const float* __restrict__ W,
                               const float* __restrict__ bias, float* __restrict__ out,
                               int R, int N, int K, int act) {
  const int idx = blockIdx.x * blockDim.x + threadIdx.x;
  if (idx >= R * N) return;
  const int r = idx / N, n = idx % N;
  const float* x = X + (long long)r * K;
  const float* wr = W + (long long)n * K;
  float s = 0.0f;
  for (int k = 0; k < K; k += 4)
    s += x[k] * wr[k] + x[k + 1] * wr[k + 1] + x[k + 2] * wr[k + 2] + x[k + 3] * wr[k + 3];
  if (bias) s += bias[n];
  if (act == 1) s = (s > 0.0f) ? s : (__expf(s) - 1.0f);   // ELU
  out[idx] = s;
}

// qs[m,e] = dot(q[m,:], wes[e,0:1024]); ks[m,e] = dot(k[m,:], wes[e,1024:2048])
__global__ void k_qsks(const float* __restrict__ qkv, const float* __restrict__ wes,
                       float* __restrict__ qs, float* __restrict__ ks) {
  const int m = blockIdx.x * blockDim.x + threadIdx.x;
  if (m >= 4096) return;
  const float* q  = qkv + (long long)m * 3072;
  const float* kk = q + 1024;
  float aq[4] = {0, 0, 0, 0}, ak[4] = {0, 0, 0, 0};
  for (int k = 0; k < 1024; ++k) {
    const float qv = q[k], kv = kk[k];
#pragma unroll
    for (int e = 0; e < 4; ++e) {
      aq[e] += qv * wes[e * 2048 + k];
      ak[e] += kv * wes[e * 2048 + 1024 + k];
    }
  }
#pragma unroll
  for (int e = 0; e < 4; ++e) { qs[m * 4 + e] = aq[e]; ks[m * 4 + e] = ak[e]; }
}

// v_t[b,d,j] = bf16(qkv[b*1024+j, 2048+d])   (tiled transpose)
__global__ void k_transpose_v(const float* __restrict__ qkv, __bf16* __restrict__ vt) {
  __shared__ float tile[32][33];
  const int b = blockIdx.z, j0 = blockIdx.x * 32, d0 = blockIdx.y * 32;
  const int tx = threadIdx.x, ty = threadIdx.y;
  for (int yy = ty; yy < 32; yy += 8)
    tile[yy][tx] = qkv[(long long)(b * 1024 + j0 + yy) * 3072 + 2048 + d0 + tx];
  __syncthreads();
  for (int yy = ty; yy < 32; yy += 8)
    vt[(long long)(b * 1024 + d0 + yy) * 1024 + j0 + tx] = f2bf(tile[tx][yy]);
}

// Softmax over adjacency: one wave32 per row (b,i); 32 j per lane.
__global__ __launch_bounds__(256) void k_attn(const int* __restrict__ adj,
                                              const int* __restrict__ types_p,
                                              const float* __restrict__ qs,
                                              const float* __restrict__ ks,
                                              __bf16* __restrict__ prob) {
  const int w = threadIdx.x >> 5, lane = threadIdx.x & 31;
  const int m = blockIdx.x * 8 + w;           // 0..4095
  const int b = m >> 10;
  const int ty = *types_p;
  const int* arow = adj + (long long)m * 1024;

  // per-row q-scores (4 edge types), loaded once
  const float q1 = qs[m * 4 + 0], q2 = qs[m * 4 + 1];
  const float q3 = qs[m * 4 + 2], q4 = qs[m * 4 + 3];

  float sv[32];
  float mx = -3.0e38f;
#pragma unroll
  for (int jj = 0; jj < 32; ++jj) {
    const int j = jj * 32 + lane;
    const int a = arow[j];
    float s;
    if (a > 0) {
      float sc = 0.0f;
      if ((ty >> a) & 1) {
        const float qsv = (a == 1) ? q1 : (a == 2) ? q2 : (a == 3) ? q3 : q4;
        sc = qsv + ks[(long long)(b * 1024 + j) * 4 + (a - 1)];
      }
      s = (sc > 0.0f) ? sc : 0.2f * sc;       // leaky(ALPHA=0.2)
    } else {
      s = -9.0e15f;                            // masked
    }
    sv[jj] = s;
    mx = fmaxf(mx, s);
  }
#pragma unroll
  for (int o = 16; o > 0; o >>= 1) mx = fmaxf(mx, __shfl_xor(mx, o, 32));

  float sum = 0.0f;
#pragma unroll
  for (int jj = 0; jj < 32; ++jj) {
    const float e = __expf(sv[jj] - mx);
    sv[jj] = e;
    sum += e;
  }
#pragma unroll
  for (int o = 16; o > 0; o >>= 1) sum += __shfl_xor(sum, o, 32);
  const float inv = 1.0f / sum;
#pragma unroll
  for (int jj = 0; jj < 32; ++jj)
    prob[(long long)m * 1024 + jj * 32 + lane] = f2bf(sv[jj] * inv);
}

// ---------------------------------------------------------------------------
// Host orchestration
// ---------------------------------------------------------------------------
extern "C" void kernel_launch(void* const* d_in, const int* in_sizes, int n_in,
                              void* d_out, int out_size, void* d_ws, size_t ws_size,
                              hipStream_t stream) {
  (void)in_sizes; (void)n_in; (void)out_size; (void)ws_size;

  const float* ents  = (const float*)d_in[0];
  const float* mask  = (const float*)d_in[1];
  const float* q_enc = (const float*)d_in[2];
  const int*   adj   = (const int*)d_in[3];
  const int*   typ   = (const int*)d_in[4];
  const float* initm = (const float*)d_in[5];
  const float* w_qin = (const float*)d_in[6];
  const float* b_qin = (const float*)d_in[7];
  const float* w_qt  = (const float*)d_in[8];
  const float* b_qt  = (const float*)d_in[9];
  const float* w_pl  = (const float*)d_in[10];
  const float* b_pl  = (const float*)d_in[11];
  const float* w_pc  = (const float*)d_in[12];
  const float* b_pc  = (const float*)d_in[13];
  const float* w_q   = (const float*)d_in[14];
  const float* b_q   = (const float*)d_in[15];
  const float* w_k   = (const float*)d_in[16];
  const float* b_k   = (const float*)d_in[17];
  const float* w_v   = (const float*)d_in[18];
  const float* b_v   = (const float*)d_in[19];
  const float* w_pk  = (const float*)d_in[20];
  const float* b_pk  = (const float*)d_in[21];
  const float* w_pv  = (const float*)d_in[22];
  const float* b_pv  = (const float*)d_in[23];
  const float* wes   = (const float*)d_in[24];
  const float* w_mu  = (const float*)d_in[25];
  const float* b_mu  = (const float*)d_in[26];
  const float* w_cb  = (const float*)d_in[27];
  const float* b_cb  = (const float*)d_in[28];

  char* base = (char*)d_ws;
  size_t off = 0;
  auto alloc = [&](size_t bytes) -> char* {
    char* p = base + off;
    off = (off + bytes + 255) & ~(size_t)255;
    return p;
  };

  // bf16 buffers
  __bf16* xjoint = (__bf16*)alloc(4096ULL * 3072 * 2);   // [x_loc | xc | pl*pc]
  __bf16* xcm    = (__bf16*)alloc(4096ULL * 2048 * 2);   // [xc | message]
  __bf16* xfin   = (__bf16*)alloc(4096ULL * 2048 * 2);   // [x_loc | x_ctx]
  __bf16* prob   = (__bf16*)alloc(4096ULL * 1024 * 2);
  __bf16* vt     = (__bf16*)alloc(4096ULL * 1024 * 2);   // v transposed [b,d,j]
  __bf16* wplB   = (__bf16*)alloc(1024ULL * 1024 * 2);
  __bf16* wpcB   = (__bf16*)alloc(1024ULL * 1024 * 2);
  __bf16* wqkvB  = (__bf16*)alloc(3072ULL * 3072 * 2);
  __bf16* wmuB   = (__bf16*)alloc(1024ULL * 2048 * 2);
  __bf16* wcbB   = (__bf16*)alloc(1024ULL * 2048 * 2);
  // f32 buffers
  float* pl    = (float*)alloc(4096ULL * 1024 * 4);
  float* pc    = (float*)alloc(4096ULL * 1024 * 4);
  float* qkv   = (float*)alloc(4096ULL * 3072 * 4);
  float* xctx  = (float*)alloc(4096ULL * 1024 * 4);
  float* qs    = (float*)alloc(4096ULL * 4 * 4);
  float* ks    = (float*)alloc(4096ULL * 4 * 4);
  float* qbase = (float*)alloc(4ULL * 1024 * 4);
  float* cmd   = (float*)alloc(4ULL * 1024 * 4);
  float* pk    = (float*)alloc(4ULL * 1024 * 4);
  float* pv    = (float*)alloc(4ULL * 1024 * 4);
  float* svec  = (float*)alloc(4ULL * 3072 * 4);
  float* bqkv  = (float*)alloc(3072ULL * 4);

  auto gemm = [&](const __bf16* A, int lda, const __bf16* B, int ldb, long long sB,
                  const float* bias, const float* scale, int sstr,
                  float* oF, int ldcF, __bf16* oB, int ldcB, int M, int N, int K) {
    GemmArgs g{A, lda, B, ldb, sB, bias, scale, sstr, oF, ldcF, oB, ldcB, K};
    k_gemm_bf16<<<dim3(N / 128, M / 128), 256, 0, stream>>>(g);
  };

  // ---- one-time setup -----------------------------------------------------
  k_cvt_bf<<<4096, 256, 0, stream>>>(w_pl, wplB, 1024LL * 1024);
  k_cvt_bf<<<4096, 256, 0, stream>>>(w_pc, wpcB, 1024LL * 1024);
  k_cvt_bf<<<4096, 256, 0, stream>>>(w_q, wqkvB, 1024LL * 3072);
  k_cvt_bf<<<4096, 256, 0, stream>>>(w_k, wqkvB + 1024LL * 3072, 1024LL * 3072);
  k_cvt_bf<<<4096, 256, 0, stream>>>(w_v, wqkvB + 2048LL * 3072, 1024LL * 3072);
  k_cvt_bf<<<4096, 256, 0, stream>>>(w_mu, wmuB, 1024LL * 2048);
  k_cvt_bf<<<4096, 256, 0, stream>>>(w_cb, wcbB, 1024LL * 2048);
  hipMemcpyAsync(bqkv,        b_q, 1024 * 4, hipMemcpyDeviceToDevice, stream);
  hipMemcpyAsync(bqkv + 1024, b_k, 1024 * 4, hipMemcpyDeviceToDevice, stream);
  hipMemcpyAsync(bqkv + 2048, b_v, 1024 * 4, hipMemcpyDeviceToDevice, stream);

  // q_base = elu(q_enc @ w_qin^T + b_qin)
  k_small_linear<<<16, 256, 0, stream>>>(q_enc, w_qin, b_qin, qbase, 4, 1024, 1024, 1);

  k_init_xloc<<<4096, 256, 0, stream>>>(ents, xjoint, xfin);
  k_init_xctx<<<4096, 256, 0, stream>>>(initm, xctx);
  k_prep_xc<<<4096, 256, 0, stream>>>(xctx, mask, xjoint, xcm);

  // pl = x_loc @ w_pl^T + b_pl   (x_loc is constant across iterations)
  gemm(xjoint, 3072, wplB, 1024, 0, b_pl, nullptr, 0,
       pl, 1024, nullptr, 0, 4096, 1024, 1024);

  // ---- 4 message-passing iterations --------------------------------------
  for (int t = 0; t < 4; ++t) {
    // cmd, pk, pv (tiny f32 linears)
    k_small_linear<<<16, 256, 0, stream>>>(qbase, w_qt + (long long)t * 1024 * 1024,
                                           b_qt + t * 1024, cmd, 4, 1024, 1024, 0);
    k_small_linear<<<16, 256, 0, stream>>>(cmd, w_pk, b_pk, pk, 4, 1024, 1024, 0);
    k_small_linear<<<16, 256, 0, stream>>>(cmd, w_pv, b_pv, pv, 4, 1024, 1024, 0);
    k_svec<<<48, 256, 0, stream>>>(pk, pv, svec);

    // pc = xc @ w_pc^T + b_pc
    gemm(xjoint + 1024, 3072, wpcB, 1024, 0, b_pc, nullptr, 0,
         pc, 1024, nullptr, 0, 4096, 1024, 1024);
    k_pack_joint<<<4096, 256, 0, stream>>>(pl, pc, xjoint);

    // fused q|k|v = x_joint @ [w_q;w_k;w_v]^T + b, epilogue *= [1|pk|pv]
    gemm(xjoint, 3072, wqkvB, 3072, 0, bqkv, svec, 3072,
         qkv, 3072, nullptr, 0, 4096, 3072, 3072);

    k_qsks<<<16, 256, 0, stream>>>(qkv, wes, qs, ks);
    k_transpose_v<<<dim3(32, 32, 4), dim3(32, 8), 0, stream>>>(qkv, vt);
    k_attn<<<512, 256, 0, stream>>>(adj, typ, qs, ks, prob);

    // message = prob @ v  (batched B), bf16 straight into xcm[:,1024:2048]
    gemm(prob, 1024, vt, 1024, 1024LL * 1024, nullptr, nullptr, 0,
         nullptr, 0, xcm + 1024, 2048, 4096, 1024, 1024);

    // x_ctx = [xc | message] @ w_mu^T + b_mu
    gemm(xcm, 2048, wmuB, 2048, 0, b_mu, nullptr, 0,
         xctx, 1024, nullptr, 0, 4096, 1024, 2048);

    // xc for next iteration (masked, bf16, both concat buffers)
    k_prep_xc<<<4096, 256, 0, stream>>>(xctx, mask, xjoint, xcm);
  }

  // ---- final projection ---------------------------------------------------
  k_fin_xctx<<<4096, 256, 0, stream>>>(xctx, xfin);
  gemm(xfin, 2048, wcbB, 2048, 0, b_cb, nullptr, 0,
       (float*)d_out, 1024, nullptr, 0, 4096, 1024, 2048);
}
